// KMeans_32950989095151
// MI455X (gfx1250) — matrix operand order repaired and checked by reading, not compiled, not run on hardware
//
#include <hip/hip_runtime.h>
#include <hip/hip_bf16.h>

// K-means step on MI455X (gfx1250, wave32, WMMA bf16).
//
// Roofline: X is 512 MB (22 us at 23.3 TB/s). GEMM is 68.7 GFLOP -> needs
// matrix cores (v_wmma_f32_16x16x32_bf16) to not dominate. argmin only needs
// the ranking of  c2[k]/2 - dot(x,c_k), so bf16 A/B with f32 accumulate is
// the precision the chip rewards. Scatter = global f32 atomics into a 1 MB
// L2-resident buffer.

typedef __attribute__((ext_vector_type(16))) __bf16 v16bf;
typedef __attribute__((ext_vector_type(8)))  float  v8f;

#define N_PTS 131072
#define DIM   1024
#define KC    256
#define LDS_STRIDE 40   // bf16 elems per centroid row chunk (32 + 8 pad -> 80B, 16B aligned, conflict-free)

// ---------------------------------------------------------------------------
// Prep: Ct[k][d] = bf16(C[d][k]);  c2h[k] = 0.5 * sum_d C[d][k]^2
// ---------------------------------------------------------------------------
__global__ void kmeans_prep(const float* __restrict__ C,
                            __bf16* __restrict__ Ct,
                            float* __restrict__ c2h) {
    const int k = blockIdx.x;          // 0..255
    const int t = threadIdx.x;         // 0..255
    float s = 0.0f;
    for (int d = t; d < DIM; d += 256) {
        float v = C[(size_t)d * KC + k];
        s += v * v;
        Ct[(size_t)k * DIM + d] = (__bf16)v;
    }
    __shared__ float red[256];
    red[t] = s;
    __syncthreads();
    for (int off = 128; off > 0; off >>= 1) {
        if (t < off) red[t] += red[t + off];
        __syncthreads();
    }
    if (t == 0) c2h[k] = 0.5f * red[0];
}

// ---------------------------------------------------------------------------
// Main: per block = 128 rows (8 waves x 16-row A tiles) x all 256 centroids.
// ---------------------------------------------------------------------------
__global__ __launch_bounds__(256) void kmeans_main(const float* __restrict__ X,
                                                   const __bf16* __restrict__ Ct,
                                                   const float* __restrict__ c2h,
                                                   float* __restrict__ sums,
                                                   float* __restrict__ counts) {
    __shared__ __bf16 ldsC[KC * LDS_STRIDE];   // 20480 B

    const int tid  = threadIdx.x;
    const int lane = tid & 31;
    const int wave = tid >> 5;
    const int half = lane >> 4;        // 0: lanes 0-15, 1: lanes 16-31
    const int l16  = lane & 15;
    const int m0   = blockIdx.x * 128 + wave * 16;

    v8f acc[16];
    #pragma unroll
    for (int t = 0; t < 16; ++t) acc[t] = (v8f){0,0,0,0,0,0,0,0};

    // A-fragment row for this lane (ISA 16-bit A layout: M = lane%16)
    const float* xrow = X + (size_t)(m0 + l16) * DIM;
    const int abase = half * 16;       // lanes<16: K 0-15; lanes>=16: K 16-31
    const int dl    = half * 16;       // B fragment K offset per half

    for (int d0 = 0; d0 < DIM; d0 += 32) {
        // ---- stage 256x32 bf16 centroid chunk into LDS (cooperative) ----
        {
            const int seg   = tid & 3;       // 16B segment within 64B row chunk
            const int kbase = tid >> 2;      // 0..63
            #pragma unroll
            for (int i = 0; i < 4; ++i) {
                const int k = i * 64 + kbase;
                const uint4 v = *reinterpret_cast<const uint4*>(
                    Ct + (size_t)k * DIM + d0 + seg * 8);
                *reinterpret_cast<uint4*>(&ldsC[k * LDS_STRIDE + seg * 8]) = v;
            }
        }
        __syncthreads();

        // ---- A fragment: 16 fp32 -> bf16, contiguous K order (k = 16*half + s) ----
        v16bf a;
        {
            const float4* p0 = reinterpret_cast<const float4*>(xrow + d0 + abase);
            const float4  u0 = p0[0], u1 = p0[1];
            const float4* p1 = reinterpret_cast<const float4*>(xrow + d0 + abase + 8);
            const float4  u2 = p1[0], u3 = p1[1];
            a[0]=(__bf16)u0.x;  a[1]=(__bf16)u0.y;  a[2]=(__bf16)u0.z;  a[3]=(__bf16)u0.w;
            a[4]=(__bf16)u1.x;  a[5]=(__bf16)u1.y;  a[6]=(__bf16)u1.z;  a[7]=(__bf16)u1.w;
            a[8]=(__bf16)u2.x;  a[9]=(__bf16)u2.y;  a[10]=(__bf16)u2.z; a[11]=(__bf16)u2.w;
            a[12]=(__bf16)u3.x; a[13]=(__bf16)u3.y; a[14]=(__bf16)u3.z; a[15]=(__bf16)u3.w;
        }

        // ---- 16 column tiles: B from LDS (32B contiguous per lane), WMMA ----
        #pragma unroll
        for (int t = 0; t < 16; ++t) {
            const __bf16* bp = &ldsC[(t * 16 + l16) * LDS_STRIDE + dl];
            v16bf b = *reinterpret_cast<const v16bf*>(bp);
            acc[t] = __builtin_amdgcn_wmma_f32_16x16x32_bf16(
                false, a, false, b, (short)0, acc[t], false, false);
        }
        __syncthreads();
    }

    // ---- per-lane half-centroid-norms for its column slots ----
    float c2r[16];
    #pragma unroll
    for (int t = 0; t < 16; ++t) c2r[t] = c2h[t * 16 + l16];

    // ---- argmin per row + fused scatter (segment sum) ----
    #pragma unroll
    for (int r = 0; r < 8; ++r) {
        // C layout: VGPR r holds row m0 + r + 8*half, col = t*16 + l16
        float bestv = c2r[0] - acc[0][r];
        int   besti = l16;
        #pragma unroll
        for (int t = 1; t < 16; ++t) {
            const float v = c2r[t] - acc[t][r];
            if (v < bestv) { bestv = v; besti = t * 16 + l16; }
        }
        // reduce across the 16 lanes of each half (xor of low 4 bits stays in half)
        #pragma unroll
        for (int m = 1; m < 16; m <<= 1) {
            const float ov = __shfl_xor(bestv, m, 32);
            const int   oi = __shfl_xor(besti, m, 32);
            if (ov < bestv || (ov == bestv && oi < besti)) { bestv = ov; besti = oi; }
        }
        const int a_lo = __shfl(besti, 0, 32);   // assignment of row m0+r
        const int a_hi = __shfl(besti, 16, 32);  // assignment of row m0+r+8

        const float* xlo = X + (size_t)(m0 + r) * DIM;
        const float* xhi = X + (size_t)(m0 + r + 8) * DIM;
        float* slo = sums + (size_t)a_lo * DIM;
        float* shi = sums + (size_t)a_hi * DIM;
        for (int d = lane; d < DIM; d += 32) {
            atomicAdd(&slo[d], xlo[d]);          // L2-resident 1 MB target
            atomicAdd(&shi[d], xhi[d]);
        }
        if (lane == 0) {
            atomicAdd(&counts[a_lo], 1.0f);
            atomicAdd(&counts[a_hi], 1.0f);
        }
    }
}

// ---------------------------------------------------------------------------
// Finalize: out[d*K + k] = sums[k*D + d] / counts[k]
// ---------------------------------------------------------------------------
__global__ void kmeans_finalize(const float* __restrict__ sums,
                                const float* __restrict__ counts,
                                float* __restrict__ out) {
    const int idx = blockIdx.x * blockDim.x + threadIdx.x;  // 0 .. D*K-1
    const int d = idx >> 8;      // K == 256
    const int k = idx & 255;
    out[idx] = sums[(size_t)k * DIM + d] / counts[k];
}

// ---------------------------------------------------------------------------
extern "C" void kernel_launch(void* const* d_in, const int* in_sizes, int n_in,
                              void* d_out, int out_size, void* d_ws, size_t ws_size,
                              hipStream_t stream) {
    const float* X = (const float*)d_in[0];   // [N, D] fp32
    const float* C = (const float*)d_in[1];   // [D, K] fp32
    float* out = (float*)d_out;               // [D, K] fp32

    // workspace layout
    float*  sums   = (float*)d_ws;                       // K*D fp32 = 1 MB
    float*  counts = sums + (size_t)KC * DIM;            // K fp32
    float*  c2h    = counts + KC;                        // K fp32
    __bf16* Ct     = (__bf16*)(c2h + KC);                // K*D bf16 = 512 KB (16B aligned)

    // sums/counts must be zero every launch (graph replays don't re-poison)
    hipMemsetAsync(d_ws, 0, ((size_t)KC * DIM + KC) * sizeof(float), stream);

    kmeans_prep<<<KC, 256, 0, stream>>>(C, Ct, c2h);
    kmeans_main<<<N_PTS / 128, 256, 0, stream>>>(X, Ct, c2h, sums, counts);
    kmeans_finalize<<<(DIM * KC) / 256, 256, 0, stream>>>(sums, counts, out);
}
